// GAT_31722628448412
// MI455X (gfx1250) — matrix operand order repaired and checked by reading, not compiled, not run on hardware
//
#include <hip/hip_runtime.h>
#include <hip/hip_bf16.h>
#include <hip/hip_fp16.h>

// ---------------------------------------------------------------------------
// GAT (2-layer) for MI455X / gfx1250, wave32 + WMMA f32_16x16x32_f16 GEMMs.
// GEMM1: one wave per 16-row M-tile, 4 accumulators (all 64 output channels),
//        A fragment converted f32->f16 once per k-step and reused 4x,
//        B fragments served from LDS-staged W1^T.
// ---------------------------------------------------------------------------

typedef __attribute__((ext_vector_type(16))) _Float16 v16h;
typedef __attribute__((ext_vector_type(8)))  float    v8f;

#define NEG_SLOPE 0.2f
#define W1S_STRIDE 144   // 128 halfs + 16 pad (288B rows: 32B-aligned, bank-staggered)

// order-preserving float -> u32 key (for atomicMax-based segment max)
__device__ __forceinline__ unsigned enc_f32(float f) {
  unsigned u = __float_as_uint(f);
  return (u & 0x80000000u) ? ~u : (u | 0x80000000u);
}
__device__ __forceinline__ float dec_f32(unsigned k) {
  unsigned u = (k & 0x80000000u) ? (k & 0x7FFFFFFFu) : ~k;
  return __uint_as_float(u);
}

__device__ __forceinline__ void edge_sd(const int* __restrict__ ei, long long E0,
                                        long long e, int& s, int& d) {
  if (e < E0) { s = ei[e]; d = ei[E0 + e]; }
  else        { s = d = (int)(e - E0); }       // self loops appended
}

// ---------------- weight prep: transpose + fp16 convert (pad W2 to 16 cols) --
__global__ void gat_prep_weights(const float* __restrict__ W1,
                                 const float* __restrict__ W2,
                                 _Float16* __restrict__ W1t,   // [64][128]
                                 _Float16* __restrict__ W2t) { // [16][64]
  for (int i = threadIdx.x; i < 64 * 128; i += blockDim.x) {
    int c = i >> 7, k = i & 127;
    W1t[i] = (_Float16)W1[k * 64 + c];
  }
  for (int i = threadIdx.x; i < 16 * 64; i += blockDim.x) {
    int c = i >> 6, k = i & 63;
    W2t[i] = (_Float16)((c < 10) ? W2[k * 10 + c] : 0.0f);
  }
}

// ---------------- GEMM1: h1[N,64] = x[N,128] @ W1 (WMMA, fp16 in / fp32 acc) -
__global__ __launch_bounds__(256) void gat_gemm1_wmma(
    const float* __restrict__ x, const _Float16* __restrict__ W1t,
    float* __restrict__ h1, int nMtiles) {
  __shared__ __align__(32) _Float16 w1s[64 * W1S_STRIDE];   // 18 KB

  // cooperative stage of W1^T (64 cols x 128 halfs) into padded LDS rows
  {
    const uint4* src = (const uint4*)W1t;         // 1024 x 16B chunks
    for (int i = threadIdx.x; i < 64 * 16; i += blockDim.x) {
      int col = i >> 4, off = i & 15;
      *((uint4*)(w1s + col * W1S_STRIDE) + off) = src[i];
    }
  }
  __syncthreads();

  const int lane  = threadIdx.x & 31;
  const int mtile = blockIdx.x * (blockDim.x >> 5) + (threadIdx.x >> 5);
  if (mtile >= nMtiles) return;

  const int row  = mtile * 16 + (lane & 15);
  const int kb   = (lane >> 4) * 16;   // A-frag K sub-base within a 32-step
  const int kgB  = (lane >> 4) * 16;   // B-frag K sub-base within a 32-step
  const int colL = lane & 15;

  const float4* xr = (const float4*)(x + (long long)row * 128);
  v8f acc[4] = {};

#pragma unroll
  for (int ks = 0; ks < 4; ++ks) {
    const int k0 = ks * 32;
    float4 f0 = xr[(k0 + kb) >> 2];
    float4 f1 = xr[(k0 + kb + 4) >> 2];
    float4 f2 = xr[(k0 + kb + 8) >> 2];
    float4 f3 = xr[(k0 + kb + 12) >> 2];
    v16h a;
    a[0]  = (_Float16)f0.x; a[1]  = (_Float16)f0.y;
    a[2]  = (_Float16)f0.z; a[3]  = (_Float16)f0.w;
    a[4]  = (_Float16)f1.x; a[5]  = (_Float16)f1.y;
    a[6]  = (_Float16)f1.z; a[7]  = (_Float16)f1.w;
    a[8]  = (_Float16)f2.x; a[9]  = (_Float16)f2.y;
    a[10] = (_Float16)f2.z; a[11] = (_Float16)f2.w;
    a[12] = (_Float16)f3.x; a[13] = (_Float16)f3.y;
    a[14] = (_Float16)f3.z; a[15] = (_Float16)f3.w;
#pragma unroll
    for (int nt = 0; nt < 4; ++nt) {
      v16h b = *(const v16h*)(w1s + (nt * 16 + colL) * W1S_STRIDE + k0 + kgB);
      acc[nt] = __builtin_amdgcn_wmma_f32_16x16x32_f16(false, a, false, b,
                                                       (short)0, acc[nt],
                                                       false, false);
    }
  }
#pragma unroll
  for (int nt = 0; nt < 4; ++nt) {
    float* outp = h1 + (long long)(mtile * 16 + (lane >> 4) * 8) * 64
                     + nt * 16 + colL;
#pragma unroll
    for (int r = 0; r < 8; ++r) outp[(long long)r * 64] = acc[nt][r];
  }
}

// ---------------- alpha1: as/ad[N,8] = einsum('nhc,hc->nh') -------------------
__global__ void gat_alpha1(const float* __restrict__ h1,
                           const float* __restrict__ a_src,
                           const float* __restrict__ a_dst,
                           float* __restrict__ as1, float* __restrict__ ad1,
                           int N) {
  int i = blockIdx.x * blockDim.x + threadIdx.x;   // n*8 + h
  if (i >= N * 8) return;
  int h = i & 7, n = i >> 3;
  const float* hp = h1 + (long long)n * 64 + h * 8;
  float s = 0.f, d = 0.f;
#pragma unroll
  for (int c = 0; c < 8; ++c) {
    s += hp[c] * a_src[h * 8 + c];
    d += hp[c] * a_dst[h * 8 + c];
  }
  as1[i] = s; ad1[i] = d;
}

// ---------------- layer1 edge kernels (thread per (edge, head)) --------------
__global__ void gat_edge_max1(const int* __restrict__ ei, long long E0,
                              long long E, const float* __restrict__ as1,
                              const float* __restrict__ ad1,
                              unsigned* __restrict__ mkey) {
  long long i = (long long)blockIdx.x * blockDim.x + threadIdx.x;
  if (i >= E * 8) return;
  int h = (int)(i & 7); long long e = i >> 3;
  int s, d; edge_sd(ei, E0, e, s, d);
  float v = as1[(long long)s * 8 + h] + ad1[(long long)d * 8 + h];
  v = (v > 0.f) ? v : NEG_SLOPE * v;
  atomicMax(&mkey[(long long)d * 8 + h], enc_f32(v));
}

__global__ void gat_decode_m(const unsigned* __restrict__ mkey,
                             float* __restrict__ m, long long n) {
  long long i = (long long)blockIdx.x * blockDim.x + threadIdx.x;
  if (i >= n) return;
  float f = dec_f32(mkey[i]);
  unsigned u = __float_as_uint(f);
  if ((u & 0x7F800000u) == 0x7F800000u) f = 0.f;   // non-finite -> 0
  m[i] = f;
}

__global__ void gat_edge_sum1(const int* __restrict__ ei, long long E0,
                              long long E, const float* __restrict__ as1,
                              const float* __restrict__ ad1,
                              const float* __restrict__ m1,
                              float* __restrict__ den1) {
  long long i = (long long)blockIdx.x * blockDim.x + threadIdx.x;
  if (i >= E * 8) return;
  int h = (int)(i & 7); long long e = i >> 3;
  int s, d; edge_sd(ei, E0, e, s, d);
  float v = as1[(long long)s * 8 + h] + ad1[(long long)d * 8 + h];
  v = (v > 0.f) ? v : NEG_SLOPE * v;
  atomicAdd(&den1[(long long)d * 8 + h], __expf(v - m1[(long long)d * 8 + h]));
}

__global__ void gat_message1(const int* __restrict__ ei, long long E0,
                             long long E, const float* __restrict__ as1,
                             const float* __restrict__ ad1,
                             const float* __restrict__ m1,
                             const float* __restrict__ den1,
                             const float* __restrict__ h1,
                             float* __restrict__ out1) {
  long long i = (long long)blockIdx.x * blockDim.x + threadIdx.x;
  if (i >= E * 8) return;
  int h = (int)(i & 7); long long e = i >> 3;
  int s, d; edge_sd(ei, E0, e, s, d);
  long long idx = (long long)d * 8 + h;
  float v = as1[(long long)s * 8 + h] + ad1[idx];
  v = (v > 0.f) ? v : NEG_SLOPE * v;
  float al = __expf(v - m1[idx]) / (den1[idx] + 1e-16f);
  const float* hs = h1 + (long long)s * 64 + h * 8;
  __builtin_prefetch(hs, 0, 0);   // global_prefetch_b8
  float* od = out1 + (long long)d * 64 + h * 8;
#pragma unroll
  for (int c = 0; c < 8; ++c) atomicAdd(&od[c], hs[c] * al);
}

// ---------------- bias + ELU (in place: out1 -> x2) --------------------------
__global__ void gat_bias_elu(float* __restrict__ out1,
                             const float* __restrict__ b1, int N) {
  int i = blockIdx.x * blockDim.x + threadIdx.x;
  if (i >= N * 64) return;
  float v = out1[i] + b1[i & 63];
  out1[i] = (v > 0.f) ? v : (__expf(v) - 1.0f);
}

// ---------------- GEMM2: h2[N,16] = x2[N,64] @ W2pad (WMMA) ------------------
__global__ __launch_bounds__(256) void gat_gemm2_wmma(
    const float* __restrict__ x2, const _Float16* __restrict__ W2t,
    float* __restrict__ h2, int nMtiles) {
  const int lane  = threadIdx.x & 31;
  const int mtile = blockIdx.x * (blockDim.x >> 5) + (threadIdx.x >> 5);
  if (mtile >= nMtiles) return;

  const int row = mtile * 16 + (lane & 15);
  const int kb  = (lane >> 4) * 16;
  const int col = lane & 15;
  const int kgB = (lane >> 4) * 16;

  const float4* xr = (const float4*)(x2 + (long long)row * 64);
  v8f acc = {};

#pragma unroll
  for (int ks = 0; ks < 2; ++ks) {
    const int k0 = ks * 32;
    float4 f0 = xr[(k0 + kb) >> 2];
    float4 f1 = xr[(k0 + kb + 4) >> 2];
    float4 f2 = xr[(k0 + kb + 8) >> 2];
    float4 f3 = xr[(k0 + kb + 12) >> 2];
    v16h a;
    a[0]  = (_Float16)f0.x; a[1]  = (_Float16)f0.y;
    a[2]  = (_Float16)f0.z; a[3]  = (_Float16)f0.w;
    a[4]  = (_Float16)f1.x; a[5]  = (_Float16)f1.y;
    a[6]  = (_Float16)f1.z; a[7]  = (_Float16)f1.w;
    a[8]  = (_Float16)f2.x; a[9]  = (_Float16)f2.y;
    a[10] = (_Float16)f2.z; a[11] = (_Float16)f2.w;
    a[12] = (_Float16)f3.x; a[13] = (_Float16)f3.y;
    a[14] = (_Float16)f3.z; a[15] = (_Float16)f3.w;
    v16h b = *(const v16h*)(W2t + (long long)col * 64 + k0 + kgB);
    acc = __builtin_amdgcn_wmma_f32_16x16x32_f16(false, a, false, b,
                                                 (short)0, acc, false, false);
  }
  float* out = h2 + (long long)(mtile * 16 + (lane >> 4) * 8) * 16 + (lane & 15);
#pragma unroll
  for (int r = 0; r < 8; ++r) out[(long long)r * 16] = acc[r];
}

// ---------------- alpha2 (H=1, C=10) -----------------------------------------
__global__ void gat_alpha2(const float* __restrict__ h2,
                           const float* __restrict__ a_src,
                           const float* __restrict__ a_dst,
                           float* __restrict__ as2, float* __restrict__ ad2,
                           int N) {
  int n = blockIdx.x * blockDim.x + threadIdx.x;
  if (n >= N) return;
  const float* hp = h2 + (long long)n * 16;
  float s = 0.f, d = 0.f;
#pragma unroll
  for (int c = 0; c < 10; ++c) { s += hp[c] * a_src[c]; d += hp[c] * a_dst[c]; }
  as2[n] = s; ad2[n] = d;
}

__global__ void gat_edge_max2(const int* __restrict__ ei, long long E0,
                              long long E, const float* __restrict__ as2,
                              const float* __restrict__ ad2,
                              unsigned* __restrict__ mkey) {
  long long e = (long long)blockIdx.x * blockDim.x + threadIdx.x;
  if (e >= E) return;
  int s, d; edge_sd(ei, E0, e, s, d);
  float v = as2[s] + ad2[d];
  v = (v > 0.f) ? v : NEG_SLOPE * v;
  atomicMax(&mkey[d], enc_f32(v));
}

__global__ void gat_edge_sum2(const int* __restrict__ ei, long long E0,
                              long long E, const float* __restrict__ as2,
                              const float* __restrict__ ad2,
                              const float* __restrict__ m2,
                              float* __restrict__ den2) {
  long long e = (long long)blockIdx.x * blockDim.x + threadIdx.x;
  if (e >= E) return;
  int s, d; edge_sd(ei, E0, e, s, d);
  float v = as2[s] + ad2[d];
  v = (v > 0.f) ? v : NEG_SLOPE * v;
  atomicAdd(&den2[d], __expf(v - m2[d]));
}

__global__ void gat_message2(const int* __restrict__ ei, long long E0,
                             long long E, const float* __restrict__ as2,
                             const float* __restrict__ ad2,
                             const float* __restrict__ m2,
                             const float* __restrict__ den2,
                             const float* __restrict__ h2,
                             float* __restrict__ out2) {
  long long e = (long long)blockIdx.x * blockDim.x + threadIdx.x;
  if (e >= E) return;
  int s, d; edge_sd(ei, E0, e, s, d);
  float v = as2[s] + ad2[d];
  v = (v > 0.f) ? v : NEG_SLOPE * v;
  float al = __expf(v - m2[d]) / (den2[d] + 1e-16f);
  const float* hs = h2 + (long long)s * 16;
  __builtin_prefetch(hs, 0, 0);
  float* od = out2 + (long long)d * 10;
#pragma unroll
  for (int c = 0; c < 10; ++c) atomicAdd(&od[c], hs[c] * al);
}

// ---------------- global mean pool + log_softmax -----------------------------
__global__ void gat_pool(const float* __restrict__ out2,
                         const int* __restrict__ batch,
                         float* __restrict__ sums, float* __restrict__ cnt,
                         int N) {
  int n = blockIdx.x * blockDim.x + threadIdx.x;
  if (n >= N) return;
  int g = batch[n];
  const float* r = out2 + (long long)n * 10;
#pragma unroll
  for (int c = 0; c < 10; ++c) atomicAdd(&sums[g * 10 + c], r[c]);
  atomicAdd(&cnt[g], 1.0f);
}

__global__ void gat_finalize(const float* __restrict__ sums,
                             const float* __restrict__ cnt,
                             const float* __restrict__ b2,
                             float* __restrict__ out, int G) {
  int g = blockIdx.x * blockDim.x + threadIdx.x;
  if (g >= G) return;
  float cn = fmaxf(cnt[g], 1.0f);
  float v[10], mx = -3.402823466e+38f;
#pragma unroll
  for (int c = 0; c < 10; ++c) { v[c] = sums[g * 10 + c] / cn + b2[c]; mx = fmaxf(mx, v[c]); }
  float se = 0.f;
#pragma unroll
  for (int c = 0; c < 10; ++c) se += __expf(v[c] - mx);
  float ls = __logf(se);
#pragma unroll
  for (int c = 0; c < 10; ++c) out[g * 10 + c] = v[c] - mx - ls;
}

// ---------------------------------------------------------------------------
extern "C" void kernel_launch(void* const* d_in, const int* in_sizes, int n_in,
                              void* d_out, int out_size, void* d_ws, size_t ws_size,
                              hipStream_t stream) {
  const float* x      = (const float*)d_in[0];
  const int*   ei     = (const int*)d_in[1];
  const int*   batch  = (const int*)d_in[2];
  const float* W1     = (const float*)d_in[3];
  const float* a_src1 = (const float*)d_in[4];
  const float* a_dst1 = (const float*)d_in[5];
  const float* b1     = (const float*)d_in[6];
  const float* W2     = (const float*)d_in[7];
  const float* a_src2 = (const float*)d_in[8];
  const float* a_dst2 = (const float*)d_in[9];
  const float* b2     = (const float*)d_in[10];
  float* out = (float*)d_out;

  const int       N  = in_sizes[0] / 128;          // 100000
  const long long E0 = in_sizes[1] / 2;            // 1600000
  const long long E  = E0 + N;                     // + self loops
  const int       G  = 512;

  // ---- carve workspace (256B-aligned sub-allocations) ----
  char* p = (char*)d_ws;
  auto alloc = [&](size_t bytes) -> void* {
    void* r = (void*)p;
    p += (bytes + 255) & ~(size_t)255;
    return r;
  };
  _Float16* W1t = (_Float16*)alloc(64 * 128 * sizeof(_Float16));
  _Float16* W2t = (_Float16*)alloc(16 * 64 * sizeof(_Float16));
  float* h1  = (float*)alloc((size_t)N * 64 * 4);
  float* as1 = (float*)alloc((size_t)N * 8 * 4);
  float* ad1 = (float*)alloc((size_t)N * 8 * 4);
  float* m1  = (float*)alloc((size_t)N * 8 * 4);
  float* h2  = (float*)alloc((size_t)N * 16 * 4);
  float* as2 = (float*)alloc((size_t)N * 4);
  float* ad2 = (float*)alloc((size_t)N * 4);
  float* m2  = (float*)alloc((size_t)N * 4);
  // contiguous zero-init region:
  char* zbeg = p;
  unsigned* mkey1 = (unsigned*)alloc((size_t)N * 8 * 4);
  float*    den1  = (float*)alloc((size_t)N * 8 * 4);
  float*    out1  = (float*)alloc((size_t)N * 64 * 4);   // becomes x2 in place
  unsigned* mkey2 = (unsigned*)alloc((size_t)N * 4);
  float*    den2  = (float*)alloc((size_t)N * 4);
  float*    out2  = (float*)alloc((size_t)N * 10 * 4);
  float*    sums  = (float*)alloc((size_t)G * 10 * 4);
  float*    cnt   = (float*)alloc((size_t)G * 4);
  size_t zbytes = (size_t)(p - zbeg);
  (void)ws_size; (void)n_in; (void)out_size;

  hipMemsetAsync(zbeg, 0, zbytes, stream);
  gat_prep_weights<<<1, 256, 0, stream>>>(W1, W2, W1t, W2t);

  // ---- layer 1 ----
  const int nMtiles = (N + 15) / 16;                       // 6250
  gat_gemm1_wmma<<<(nMtiles + 7) / 8, 256, 0, stream>>>(x, W1t, h1, nMtiles);
  gat_alpha1<<<(N * 8 + 255) / 256, 256, 0, stream>>>(h1, a_src1, a_dst1, as1, ad1, N);
  {
    long long t = E * 8;
    int blocks = (int)((t + 255) / 256);
    gat_edge_max1<<<blocks, 256, 0, stream>>>(ei, E0, E, as1, ad1, mkey1);
    gat_decode_m<<<(int)(((long long)N * 8 + 255) / 256), 256, 0, stream>>>(mkey1, m1, (long long)N * 8);
    gat_edge_sum1<<<blocks, 256, 0, stream>>>(ei, E0, E, as1, ad1, m1, den1);
    gat_message1<<<blocks, 256, 0, stream>>>(ei, E0, E, as1, ad1, m1, den1, h1, out1);
  }
  gat_bias_elu<<<(N * 64 + 255) / 256, 256, 0, stream>>>(out1, b1, N);

  // ---- layer 2 ----
  gat_gemm2_wmma<<<(nMtiles + 7) / 8, 256, 0, stream>>>(out1, W2t, h2, nMtiles);
  gat_alpha2<<<(N + 255) / 256, 256, 0, stream>>>(h2, a_src2, a_dst2, as2, ad2, N);
  {
    int blocks = (int)((E + 255) / 256);
    gat_edge_max2<<<blocks, 256, 0, stream>>>(ei, E0, E, as2, ad2, mkey2);
    gat_decode_m<<<(int)(((long long)N + 255) / 256), 256, 0, stream>>>(mkey2, m2, (long long)N);
    gat_edge_sum2<<<blocks, 256, 0, stream>>>(ei, E0, E, as2, ad2, m2, den2);
    gat_message2<<<blocks, 256, 0, stream>>>(ei, E0, E, as2, ad2, m2, den2, h2, out2);
  }

  // ---- pool + log_softmax ----
  gat_pool<<<(N + 255) / 256, 256, 0, stream>>>(out2, batch, sums, cnt, N);
  gat_finalize<<<(G + 255) / 256, 256, 0, stream>>>(sums, cnt, b2, out, G);
}
